// CorrespondenceFeatGenerationArch_47330539602238
// MI455X (gfx1250) — matrix operand order repaired and checked by reading, not compiled, not run on hardware
//
#include <hip/hip_runtime.h>

typedef __attribute__((ext_vector_type(16))) _Float16 v16h;
typedef __attribute__((ext_vector_type(8)))  _Float16 v8h;
typedef __attribute__((ext_vector_type(8)))  float    v8f;

#define C_DIM 256
#define HW    4096   // 64*64 pixels
#define WO    62
#define NP    3844   // 62*62 patch positions

#define BK      64   // K-step per staged tile
#define LDA     72   // padded row stride (halves): 144B = 9*16B -> b128-aligned, bank-conflict-free
#define LDB     72

// ---------------------------------------------------------------- zero output
__global__ __launch_bounds__(256) void zero_kernel(float* __restrict__ out, int n) {
    int i = blockIdx.x * 256 + threadIdx.x;
    if (i < n) out[i] = 0.0f;
}

// ------------------------------------------------- per-pixel L2 normalization
// src: (256, 64, 64) f32   ->   dst: pixel-major (4096, 256) f16
__global__ __launch_bounds__(256) void normalize_kernel(
    const float* __restrict__ f1, const float* __restrict__ f2,
    _Float16* __restrict__ An, _Float16* __restrict__ Bn) {
    int gid = blockIdx.x * 256 + threadIdx.x;     // 0..8191 = [batch][pixel]
    int pix = gid & (HW - 1);
    int b   = gid >> 12;
    const float* src = (blockIdx.y == 0 ? f1 : f2) + (size_t)b * C_DIM * HW + pix;
    _Float16*    dst = (blockIdx.y == 0 ? An : Bn) + (size_t)gid * C_DIM;

    float ss = 0.0f;
#pragma unroll 4
    for (int c = 0; c < C_DIM; ++c) { float v = src[(size_t)c * HW]; ss += v * v; }
    float inv = rsqrtf(fmaxf(ss, 1e-24f));   // matches f / max(||f||, 1e-12)

    for (int c0 = 0; c0 < C_DIM; c0 += 8) {
        v8h h;
#pragma unroll
        for (int e = 0; e < 8; ++e) h[e] = (_Float16)(src[(size_t)(c0 + e) * HW] * inv);
        *reinterpret_cast<v8h*>(dst + c0) = h;
    }
}

// --------------------------------------------------------- D = Fin^T * Fref
// A (input feats) pixel-major (4096 x 256) f16, B (ref feats) same layout.
// D[m][n] = sum_k A[m][k] * B[n][k];  M=N=4096, K=256.
// Workgroup tile 128(M) x 64(N); 8 waves; wave = 2x2 tiles of 16x16.
// Double-buffered async-to-LDS staging, BK=64 (two k=32 WMMA substeps).
__global__ __launch_bounds__(256) void gemm_kernel(
    const _Float16* __restrict__ A, const _Float16* __restrict__ B,
    float* __restrict__ D) {
    __shared__ alignas(128) _Float16 lA[2][128 * LDA];  // 2 x 18 KB
    __shared__ alignas(128) _Float16 lB[2][64 * LDB];   // 2 x  9 KB

    const int t    = threadIdx.x;
    const int lane = t & 31;
    const int wv   = t >> 5;
    const int m0   = (wv & 3) * 32;   // wave M offset within tile
    const int n0   = (wv >> 2) * 32;  // wave N offset within tile
    const int wgM  = blockIdx.y;
    const int wgN  = blockIdx.x;
    const int hh   = lane >> 4;       // half-wave select
    const int l15  = lane & 15;

    v8f acc[2][2] = {};

    // ---- async stage of one BK=64 tile pair into buffer `buf` (6 b128 / thread)
    auto issue = [&](int buf, int kk) {
        // A tile: 128 rows x 64 halves = 1024 x 16B chunks, 4 per thread
#pragma unroll
        for (int cc = 0; cc < 4; ++cc) {
            int c = t + cc * 256;
            int row = c >> 3, q = (c & 7) * 8;
            unsigned long long ga = (unsigned long long)(uintptr_t)
                (A + (size_t)(wgM * 128 + row) * C_DIM + kk + q);
            unsigned la = (unsigned)(uintptr_t)(&lA[buf][row * LDA + q]);
            asm volatile("global_load_async_to_lds_b128 %0, %1, off"
                         :: "v"(la), "v"(ga) : "memory");
        }
        // B tile: 64 rows x 64 halves = 512 x 16B chunks, 2 per thread
#pragma unroll
        for (int cc = 0; cc < 2; ++cc) {
            int c = t + cc * 256;
            int row = c >> 3, q = (c & 7) * 8;
            unsigned long long ga = (unsigned long long)(uintptr_t)
                (B + (size_t)(wgN * 64 + row) * C_DIM + kk + q);
            unsigned la = (unsigned)(uintptr_t)(&lB[buf][row * LDB + q]);
            asm volatile("global_load_async_to_lds_b128 %0, %1, off"
                         :: "v"(la), "v"(ga) : "memory");
        }
    };

    issue(0, 0);   // prologue: prefetch first tile

#pragma unroll
    for (int step = 0; step < C_DIM / BK; ++step) {
        const int cur = step & 1;
        __syncthreads();                    // everyone done reading buf cur^1
        if (step < C_DIM / BK - 1) {
            issue(cur ^ 1, (step + 1) * BK);   // prefetch next tile (overlaps compute)
            // in-order ASYNCcnt: <=6 outstanding  =>  current tile's 6 are complete
            asm volatile("s_wait_asynccnt 0x6" ::: "memory");
        } else {
            asm volatile("s_wait_asynccnt 0x0" ::: "memory");
        }
        __syncthreads();                    // current tile visible to all waves

#pragma unroll
        for (int ks = 0; ks < BK; ks += 32) {
            // A 16x32 frag: lane row m=(lane&15); elems 0..7 -> k=h*8+i, 8..15 -> k=16+h*8+i
            v16h af[2];
#pragma unroll
            for (int mt = 0; mt < 2; ++mt) {
                const _Float16* base = &lA[cur][(m0 + mt * 16 + l15) * LDA + ks + hh * 8];
                v8h lo = *reinterpret_cast<const v8h*>(base);
                v8h hi = *reinterpret_cast<const v8h*>(base + 16);
#pragma unroll
                for (int e = 0; e < 8; ++e) { af[mt][e] = lo[e]; af[mt][8 + e] = hi[e]; }
            }
            // B 32x16 frag: lane col n=(lane&15); elems 0..7 -> k=h*8+i, 8..15 -> k=16+h*8+i
            v16h bf[2];
#pragma unroll
            for (int nt = 0; nt < 2; ++nt) {
                const _Float16* base = &lB[cur][(n0 + nt * 16 + l15) * LDB + ks + hh * 8];
                v8h lo = *reinterpret_cast<const v8h*>(base);
                v8h hi = *reinterpret_cast<const v8h*>(base + 16);
#pragma unroll
                for (int e = 0; e < 8; ++e) { bf[nt][e] = lo[e]; bf[nt][8 + e] = hi[e]; }
            }
#pragma unroll
            for (int mt = 0; mt < 2; ++mt)
#pragma unroll
                for (int nt = 0; nt < 2; ++nt)
                    acc[mt][nt] = __builtin_amdgcn_wmma_f32_16x16x32_f16(
                        false, af[mt], false, bf[nt], (short)0, acc[mt][nt], false, false);
        }
    }

    // ---- store: C/D layout: VGPR v -> M = v + 8*hh, N = lane&15
#pragma unroll
    for (int mt = 0; mt < 2; ++mt)
        for (int nt = 0; nt < 2; ++nt)
#pragma unroll
            for (int v = 0; v < 8; ++v) {
                int mrow = wgM * 128 + m0 + mt * 16 + hh * 8 + v;
                int ncol = wgN * 64 + n0 + nt * 16 + l15;
                D[(size_t)mrow * HW + ncol] = acc[mt][nt][v];
            }
}

// ---------------------- 9-tap patch sum over D, argmax, flow + 9 shifted writes
__global__ __launch_bounds__(256) void argmax_flow_kernel(
    const float* __restrict__ D, float* __restrict__ out) {
    __shared__ float sv[256];
    __shared__ int   si[256];

    int pos = blockIdx.x;
    int yi = pos / WO, xi = pos % WO;

    const float* p[9];
#pragma unroll
    for (int dy = 0; dy < 3; ++dy)
#pragma unroll
        for (int dx = 0; dx < 3; ++dx)
            p[dy * 3 + dx] = D + (size_t)((yi + dy) * 64 + (xi + dx)) * HW + (dy * 64 + dx);

    int t = threadIdx.x;
    float best = -3.4e38f;
    int   bidx = 0x7fffffff;
    for (int r = t; r < NP; r += 256) {
        int yr = r / WO, xr = r - yr * WO;
        int base = yr * 64 + xr;
        float val = 0.0f;
#pragma unroll
        for (int s = 0; s < 9; ++s) val += p[s][base];
        if (val > best) { best = val; bidx = r; }   // first-max per residue class
    }
    sv[t] = best; si[t] = bidx;
    __syncthreads();
    for (int s = 128; s > 0; s >>= 1) {
        if (t < s) {
            float ov = sv[t + s]; int oi = si[t + s];
            if (ov > sv[t] || (ov == sv[t] && oi < si[t])) { sv[t] = ov; si[t] = oi; }
        }
        __syncthreads();
    }
    int idx = si[0];
    float fw = (float)(idx % WO) - (float)xi;
    float fh = (float)(idx / WO) - (float)yi;
    if (t < 9) {                        // shift s=(i,j): out[s][yi+i][xi+j] = flow
        int i = t / 3, j = t % 3;
        int h = yi + i, w = xi + j;
        float* q = out + ((size_t)(t * 64 + h) * 64 + w) * 2;
        q[0] = fw; q[1] = fh;
    }
}

// ------------------------------------------------------------------- launcher
extern "C" void kernel_launch(void* const* d_in, const int* in_sizes, int n_in,
                              void* d_out, int out_size, void* d_ws, size_t ws_size,
                              hipStream_t stream) {
    const float* f1 = (const float*)d_in[0];   // dense_features1 (2,256,64,64)
    const float* f2 = (const float*)d_in[1];   // dense_features2 (2,256,64,64)
    float* out = (float*)d_out;                // (2,9,64,64,2)

    char* ws = (char*)d_ws;
    _Float16* An = (_Float16*)ws;                        // 2*4096*256 f16 = 4 MiB
    _Float16* Bn = (_Float16*)(ws + (size_t)(4u << 20)); // 4 MiB
    float*    D  = (float*)(ws + (size_t)(8u << 20));    // 4096*4096 f32 = 64 MiB (reused per batch)

    zero_kernel<<<576, 256, 0, stream>>>(out, 2 * 9 * 64 * 64 * 2);
    normalize_kernel<<<dim3(32, 2), 256, 0, stream>>>(f1, f2, An, Bn);

    for (int b = 0; b < 2; ++b) {
        gemm_kernel<<<dim3(64, 32), 256, 0, stream>>>(
            An + (size_t)b * HW * C_DIM, Bn + (size_t)b * HW * C_DIM, D);
        argmax_flow_kernel<<<NP, 256, 0, stream>>>(
            D, out + (size_t)b * 9 * 64 * 64 * 2);
    }
    (void)in_sizes; (void)n_in; (void)out_size; (void)ws_size;
}
